// TransformerCNNNet_5798205849643
// MI455X (gfx1250) — matrix operand order repaired and checked by reading, not compiled, not run on hardware
//
#include <hip/hip_runtime.h>

// ===========================================================================
// MI455X (gfx1250) TransformerCNNNet forward. All heavy math via
// v_wmma_f32_16x16x32_f16 on pre-converted f16 operands (f32 accumulate).
// GEMM/conv use 32x64 wave tiles: 8 WMMAs per 12 b128 fragment loads.
// ===========================================================================

typedef __attribute__((ext_vector_type(16))) _Float16 v16h;
typedef __attribute__((ext_vector_type(8)))  _Float16 v8h;
typedef __attribute__((ext_vector_type(8)))  float    v8f;

static __device__ __forceinline__ v8f wmma16(v16h a, v16h b, v8f c) {
  return __builtin_amdgcn_wmma_f32_16x16x32_f16(false, a, false, b, (short)0, c,
                                                false, false);
}
static __device__ __forceinline__ v16h mk16(v8h lo, v8h hi) {
  v16h r;
#pragma unroll
  for (int i = 0; i < 8; ++i) { r[i] = lo[i]; r[8 + i] = hi[i]; }
  return r;
}

#define BQ   8
#define SQ   2048
#define DQ   256

// ---------------------------------------------------------------------------
__global__ void __launch_bounds__(256) k_cvt(const float* __restrict__ in,
                                             _Float16* __restrict__ out, int n) {
  int i = blockIdx.x * 256 + threadIdx.x;
  if (i < n) out[i] = (_Float16)in[i];
}

// ---------------------------------------------------------------------------
__global__ void __launch_bounds__(256) k_embed(const float* __restrict__ x,
                                               const float* __restrict__ ew,
                                               const float* __restrict__ eb,
                                               float* __restrict__ out,
                                               _Float16* __restrict__ out16) {
  int idx = blockIdx.x * 256 + threadIdx.x;   // over B*S*D = 4194304
  int d = idx & 255;
  int s = (idx >> 8) & 2047;
  int b = idx >> 19;
  const float* xe = x + ((size_t)b * SQ + s) * 4;
  const float* w  = ew + (size_t)d * 4;
  float v = eb[d] + xe[0]*w[0] + xe[1]*w[1] + xe[2]*w[2] + xe[3]*w[3];
  int p = d >> 1;
  float freq = __expf(-(float)(2 * p) * (9.210340371976184f / 256.0f));
  float ang  = (float)s * freq;
  v += (d & 1) ? __cosf(ang) : __sinf(ang);
  out[idx]   = v;
  out16[idx] = (_Float16)v;
}

// ---------------------------------------------------------------------------
// WMMA GEMM: C[M,N] = act(A16[M,K] @ W16[N,K]^T + bias)
// Wave tile 32x64: 2 M-subtiles x 4 N-subtiles, B fragments reused across M.
// Cf (f32) and C16 (f16) outputs each optional. M%32==0, N%64==0, K%32==0.
// ---------------------------------------------------------------------------
__global__ void __launch_bounds__(256) k_gemm(const _Float16* __restrict__ A16,
                                              const _Float16* __restrict__ W16,
                                              const float* __restrict__ bias,
                                              float* __restrict__ Cf,
                                              _Float16* __restrict__ C16,
                                              int M, int N, int K, int relu) {
  int lane = threadIdx.x & 31;
  int wv   = threadIdx.x >> 5;
  int tilesN = N >> 6;
  int tile = blockIdx.x * 8 + wv;
  if (tile >= (M >> 5) * tilesN) return;     // whole-wave exit
  int tm = tile / tilesN, tn = tile % tilesN;
  int m0 = tm << 5, n0 = tn << 6;
  int hh = lane >> 4;
  int r  = lane & 15;

  v8f z = {0.f,0.f,0.f,0.f,0.f,0.f,0.f,0.f};
  v8f acc[2][4];
#pragma unroll
  for (int mt = 0; mt < 2; ++mt)
#pragma unroll
    for (int jj = 0; jj < 4; ++jj) acc[mt][jj] = z;

  const _Float16* arow0 = A16 + (size_t)(m0 + r) * K + hh * 8;
  const _Float16* arow1 = arow0 + (size_t)16 * K;
  const _Float16* wr0   = W16 + (size_t)(n0 + r) * K + hh * 8;
  for (int k0 = 0; k0 < K; k0 += 32) {
    __builtin_prefetch(arow0 + k0 + 32, 0, 0);
    // phase 1: all fragment loads into distinct regs (clause-able)
    v8h a0l = *(const v8h*)(arow0 + k0);
    v8h a0h = *(const v8h*)(arow0 + k0 + 16);
    v8h a1l = *(const v8h*)(arow1 + k0);
    v8h a1h = *(const v8h*)(arow1 + k0 + 16);
    v8h b_lo[4], b_hi[4];
#pragma unroll
    for (int jj = 0; jj < 4; ++jj) {
      const _Float16* wrow = wr0 + (size_t)(jj * 16) * K + k0;
      b_lo[jj] = *(const v8h*)(wrow);
      b_hi[jj] = *(const v8h*)(wrow + 16);
    }
    // phase 2: 8 WMMAs, B reused across both M-subtiles
    v16h av0 = mk16(a0l, a0h);
    v16h av1 = mk16(a1l, a1h);
#pragma unroll
    for (int jj = 0; jj < 4; ++jj) {
      v16h bv = mk16(b_lo[jj], b_hi[jj]);
      acc[0][jj] = wmma16(av0, bv, acc[0][jj]);
      acc[1][jj] = wmma16(av1, bv, acc[1][jj]);
    }
  }
#pragma unroll
  for (int mt = 0; mt < 2; ++mt) {
#pragma unroll
    for (int jj = 0; jj < 4; ++jj) {
      int col = n0 + jj * 16 + r;
      float bs = bias[col];
#pragma unroll
      for (int rr = 0; rr < 8; ++rr) {
        float v = acc[mt][jj][rr] + bs;
        if (relu) v = fmaxf(v, 0.f);
        size_t o = (size_t)(m0 + mt * 16 + rr + 8 * hh) * N + col;
        if (Cf)  Cf[o]  = v;
        if (C16) C16[o] = (_Float16)v;
      }
    }
  }
}

// ---------------------------------------------------------------------------
// Flash attention over f16 qkv [B,S,768]; writes f16 out [B,S,256].
// ---------------------------------------------------------------------------
#define AW 4
__global__ void __launch_bounds__(128) k_attn(const _Float16* __restrict__ qkv,
                                              _Float16* __restrict__ out) {
  __shared__ float st[AW][16 * 32];
  __shared__ __align__(16) _Float16 stp[AW][16 * 32];
  __shared__ float mrow[AW][16], lrow[AW][16], fac[AW][16];
  int lane = threadIdx.x & 31;
  int w    = threadIdx.x >> 5;
  int tile = blockIdx.x * AW + w;             // 8192 tiles (exact grid)
  int qb  = tile & 127;
  int bh  = tile >> 7;
  int h32 = bh & 7;
  int b   = bh >> 3;
  int q0  = qb * 16;
  const _Float16* base = qkv + (size_t)b * SQ * 768;
  int hh = lane >> 4, r = lane & 15;
  const float sc = 0.17677669529663687f;      // 1/sqrt(32)

  const v8h* qp = (const v8h*)(base + (size_t)(q0 + r) * 768 + h32 * 32 + hh * 16);
  v16h aq = mk16(qp[0], qp[1]);

  v8f z = {0.f,0.f,0.f,0.f,0.f,0.f,0.f,0.f};
  v8f o0 = z, o1 = z;
  if (lane < 16) { mrow[w][lane] = -1e30f; lrow[w][lane] = 0.f; }
  __syncthreads();

  for (int j = 0; j < SQ; j += 32) {
    const _Float16* kp0 = base + (size_t)(j + r) * 768 + 256 + h32 * 32 + hh * 16;
    const _Float16* kp1 = kp0 + (size_t)16 * 768;
    v8h k0a = *(const v8h*)kp0, k0b = *(const v8h*)(kp0 + 8);
    v8h k1a = *(const v8h*)kp1, k1b = *(const v8h*)(kp1 + 8);
    v8f s0 = wmma16(aq, mk16(k0a, k0b), z);
    v8f s1 = wmma16(aq, mk16(k1a, k1b), z);
#pragma unroll
    for (int rr = 0; rr < 8; ++rr) {          // stage scaled scores
      st[w][(rr + 8 * hh) * 32 + r]      = s0[rr] * sc;
      st[w][(rr + 8 * hh) * 32 + 16 + r] = s1[rr] * sc;
    }
    __syncthreads();
    if (lane < 16) {                          // online softmax, one q-row/lane
      float m_old = mrow[w][lane];
      float mj = m_old;
      float* row = &st[w][lane * 32];
#pragma unroll
      for (int t = 0; t < 32; ++t) mj = fmaxf(mj, row[t]);
      float f = __expf(m_old - mj);
      float ls = 0.f;
#pragma unroll
      for (int t = 0; t < 32; ++t) {
        float p = __expf(row[t] - mj);
        stp[w][lane * 32 + t] = (_Float16)p;
        ls += p;
      }
      mrow[w][lane] = mj;
      lrow[w][lane] = lrow[w][lane] * f + ls;
      fac[w][lane]  = f;
    }
    __syncthreads();
#pragma unroll
    for (int rr = 0; rr < 8; ++rr) {
      float f = fac[w][rr + 8 * hh];
      o0[rr] *= f; o1[rr] *= f;
    }
    const v8h* pp = (const v8h*)&stp[w][r * 32 + hh * 16];
    v16h ap = mk16(pp[0], pp[1]);
    _Float16 va[16], vb[16];
    {
      const _Float16* vp = base + (size_t)(j + hh * 16) * 768 + 512 + h32 * 32;
#pragma unroll
      for (int i = 0; i < 16; ++i) {
        const _Float16* vr = vp + (size_t)i * 768;
        va[i] = vr[r];
        vb[i] = vr[16 + r];
      }
    }
    v16h bv0, bv1;
#pragma unroll
    for (int i = 0; i < 16; ++i) { bv0[i] = va[i]; bv1[i] = vb[i]; }
    o0 = wmma16(ap, bv0, o0);
    o1 = wmma16(ap, bv1, o1);
    __syncthreads();
  }
#pragma unroll
  for (int rr = 0; rr < 8; ++rr) {
    float inv = 1.f / lrow[w][rr + 8 * hh];
    int q = q0 + rr + 8 * hh;
    _Float16* orow = out + ((size_t)(b * SQ + q)) * DQ + h32 * 32;
    orow[r]      = (_Float16)(o0[rr] * inv);
    orow[16 + r] = (_Float16)(o1[rr] * inv);
  }
}

// ---------------------------------------------------------------------------
__global__ void __launch_bounds__(256) k_addln(const float* __restrict__ x,
                                               const float* __restrict__ rsd,
                                               const float* __restrict__ g,
                                               const float* __restrict__ bb,
                                               float* __restrict__ out,
                                               _Float16* __restrict__ out16) {
  __shared__ float red[256];
  int d = threadIdx.x;
  size_t idx = (size_t)blockIdx.x * 256 + d;
  float t = x[idx] + rsd[idx];
  red[d] = t; __syncthreads();
  for (int s2 = 128; s2 > 0; s2 >>= 1) { if (d < s2) red[d] += red[d + s2]; __syncthreads(); }
  float mean = red[0] * (1.f / 256.f);
  __syncthreads();
  float c = t - mean;
  red[d] = c * c; __syncthreads();
  for (int s2 = 128; s2 > 0; s2 >>= 1) { if (d < s2) red[d] += red[d + s2]; __syncthreads(); }
  float var = red[0] * (1.f / 256.f);
  float v = c * rsqrtf(var + 1e-5f) * g[d] + bb[d];
  out[idx]   = v;
  out16[idx] = (_Float16)v;
}

// [B,S,D] -> [B,D,S], f32 + f16
__global__ void __launch_bounds__(256) k_transpose(const float* __restrict__ in,
                                                   float* __restrict__ out,
                                                   _Float16* __restrict__ out16) {
  int idx = blockIdx.x * 256 + threadIdx.x;
  int d = idx & 255;
  int s = (idx >> 8) & 2047;
  int b = idx >> 19;
  float v = in[idx];
  size_t o = ((size_t)b * DQ + d) * SQ + s;
  out[o]   = v;
  out16[o] = (_Float16)v;
}

// ---------------------------------------------------------------------------
// Conv1d (implicit im2col) + folded BN via WMMA; f16 input/weights.
// Wave tile 32co x 64l: the bounds-checked gather feeds 2 weight tiles.
// ---------------------------------------------------------------------------
__global__ void __launch_bounds__(256) k_conv_bn(const _Float16* __restrict__ X16,
    const _Float16* __restrict__ W16, const float* __restrict__ cb,
    const float* __restrict__ bng, const float* __restrict__ bnb,
    const float* __restrict__ bnm, const float* __restrict__ bnv,
    float* __restrict__ Y, _Float16* __restrict__ Y16,
    int Bc, int Cin, int Lin, int Cout, int Lout, int KT, int PAD, int relu) {
  int lane = threadIdx.x & 31;
  int wv   = threadIdx.x >> 5;
  int tilesL = (Lout + 63) >> 6;
  int tilesC = Cout >> 5;
  long tilesPerB = (long)tilesL * tilesC;
  long tile = (long)blockIdx.x * 8 + wv;
  if (tile >= (long)Bc * tilesPerB) return;
  int b  = (int)(tile / tilesPerB);
  long tr = tile % tilesPerB;
  int tc = (int)(tr / tilesL), tl = (int)(tr % tilesL);
  int c0 = tc << 5, l0 = tl << 6;
  int hh = lane >> 4, r = lane & 15;
  int Ktot = Cin * KT;

  v8f z = {0.f,0.f,0.f,0.f,0.f,0.f,0.f,0.f};
  v8f acc[2][4];
#pragma unroll
  for (int mt = 0; mt < 2; ++mt)
#pragma unroll
    for (int jj = 0; jj < 4; ++jj) acc[mt][jj] = z;

  const _Float16* xb  = X16 + (size_t)b * Cin * Lin;
  const _Float16* wr0 = W16 + (size_t)(c0 + r) * Ktot + hh * 16;
  const _Float16* wr1 = wr0 + (size_t)16 * Ktot;
  for (int k0 = 0; k0 < Ktot; k0 += 32) {
    // phase 1: stage B fragments (im2col gather, shared by both co-subtiles)
    _Float16 bt[4][16];
#pragma unroll
    for (int jj = 0; jj < 4; ++jj) {
      int n = l0 + jj * 16 + r;
      if (KT == 8) {
        int ci0 = (k0 + hh * 16) >> 3;
        const _Float16* p0 = xb + (size_t)ci0 * Lin + (n - PAD);
        bool safe = (l0 + jj * 16 >= PAD) && (l0 + jj * 16 + 22 - PAD < Lin);
        if (safe) {
#pragma unroll
          for (int t = 0; t < 8; ++t) { bt[jj][t] = p0[t]; bt[jj][8 + t] = p0[Lin + t]; }
        } else {
#pragma unroll
          for (int i = 0; i < 16; ++i) {
            int t = i & 7;
            int pos = n - PAD + t;
            _Float16 v = (_Float16)0.f;
            if (pos >= 0 && pos < Lin) v = xb[(size_t)(ci0 + (i >> 3)) * Lin + pos];
            bt[jj][i] = v;
          }
        }
      } else {  // KT == 1 (pointwise shortcut conv)
        int nc = n < Lin ? n : Lin - 1;
#pragma unroll
        for (int i = 0; i < 16; ++i)
          bt[jj][i] = xb[(size_t)(k0 + hh * 16 + i) * Lin + nc];
      }
    }
    // phase 2: A fragments (aligned b128) + 8 WMMAs
    const v8h* ap0 = (const v8h*)(wr0 + k0);
    const v8h* ap1 = (const v8h*)(wr1 + k0);
    v16h av0 = mk16(ap0[0], ap0[1]);
    v16h av1 = mk16(ap1[0], ap1[1]);
#pragma unroll
    for (int jj = 0; jj < 4; ++jj) {
      v16h bv;
#pragma unroll
      for (int i = 0; i < 16; ++i) bv[i] = bt[jj][i];
      acc[0][jj] = wmma16(av0, bv, acc[0][jj]);
      acc[1][jj] = wmma16(av1, bv, acc[1][jj]);
    }
  }
  float* yb = Y + (size_t)b * Cout * Lout;
  _Float16* yb16 = Y16 ? Y16 + (size_t)b * Cout * Lout : (Y16);
#pragma unroll
  for (int mt = 0; mt < 2; ++mt) {
#pragma unroll
    for (int rr = 0; rr < 8; ++rr) {
      int co = c0 + mt * 16 + rr + 8 * hh;
      float scale = bng[co] * rsqrtf(bnv[co] + 1e-5f);
      float shift = bnb[co] - bnm[co] * scale;
      float bias  = cb[co];
#pragma unroll
      for (int jj = 0; jj < 4; ++jj) {
        int col = l0 + jj * 16 + r;
        if (col < Lout) {
          float v = (acc[mt][jj][rr] + bias) * scale + shift;
          if (relu) v = fmaxf(v, 0.f);
          size_t o = (size_t)co * Lout + col;
          yb[o] = v;
          if (Y16) yb16[o] = (_Float16)v;
        }
      }
    }
  }
}

__global__ void __launch_bounds__(256) k_pool(const float* __restrict__ X,
                                              float* __restrict__ Y,
                                              int BC, int Lin, int Lout) {
  int i = blockIdx.x * 256 + threadIdx.x;
  if (i >= BC * Lout) return;
  int bc = i / Lout, l = i % Lout;
  const float* p = X + (size_t)bc * Lin + 2 * l;
  Y[i] = fmaxf(p[0], p[1]);
}

__global__ void __launch_bounds__(256) k_addrelu(const float* __restrict__ Aa, int La,
                                                 const float* __restrict__ Rr, int Lr,
                                                 float* __restrict__ Y,
                                                 _Float16* __restrict__ Y16,
                                                 int BC, int m) {
  int i = blockIdx.x * 256 + threadIdx.x;
  if (i >= BC * m) return;
  int bc = i / m, l = i % m;
  float v = fmaxf(Aa[(size_t)bc * La + l] + Rr[(size_t)bc * Lr + l], 0.f);
  Y[i] = v;
  if (Y16) Y16[i] = (_Float16)v;
}

__global__ void __launch_bounds__(256) k_mean(const float* __restrict__ X,
                                              float* __restrict__ Y, int BC, int L) {
  int i = blockIdx.x * 256 + threadIdx.x;
  if (i >= BC) return;
  const float* p = X + (size_t)i * L;
  float s = 0.f;
  for (int l = 0; l < L; ++l) s += p[l];
  Y[i] = s / (float)L;
}

__global__ void __launch_bounds__(256) k_fc(const float* __restrict__ In,
                                            const float* __restrict__ W,
                                            const float* __restrict__ bb,
                                            float* __restrict__ Out,
                                            int M, int N, int K, int relu) {
  int i = blockIdx.x * 256 + threadIdx.x;
  if (i >= M * N) return;
  int m = i / N, n = i % N;
  const float* a = In + (size_t)m * K;
  const float* w = W + (size_t)n * K;
  float s = bb[n];
  for (int k = 0; k < K; ++k) s += a[k] * w[k];
  Out[i] = relu ? fmaxf(s, 0.f) : s;
}

// ===========================================================================
// Host orchestration. Input order = setup_inputs() insertion order.
// ws layout: 8 f32 slots (4.4M floats each) + f16 arena (~29.6M halves).
// ===========================================================================
extern "C" void kernel_launch(void* const* d_in, const int* in_sizes, int n_in,
                              void* d_out, int out_size, void* d_ws, size_t ws_size,
                              hipStream_t stream) {
  (void)in_sizes; (void)n_in; (void)out_size; (void)ws_size;
  auto F = [&](int i) { return (const float*)d_in[i]; };

  float* ws = (float*)d_ws;
  const size_t SLOT = 4400000;
  float* P[8];
  for (int i = 0; i < 8; ++i) P[i] = ws + SLOT * i;
  _Float16* HA = (_Float16*)(ws + 8 * SLOT);   // f16 arena

  _Float16* hc1w0 = HA + 0;            // 256*256*8
  _Float16* hc2w0 = HA + 524288;
  _Float16* hc1w1 = HA + 1048576;      // 128*256*8
  _Float16* hc2w1 = HA + 1310720;      // 128*128*8
  _Float16* hscw1 = HA + 1441792;      // 128*256*1
  _Float16* hc1w2 = HA + 1474560;
  _Float16* hc2w2 = HA + 1605632;
  _Float16* hWt   = HA + 1736704;      // 2 layers x 1310720
  _Float16* hX    = HA + 4358144;      // 16384*256
  _Float16* hQKV  = HA + 8552448;      // 16384*768
  _Float16* hAo   = HA + 21135360;     // 16384*256
  _Float16* hH    = HA + 25329664;     // 2048*2048
  _Float16* hC0 = hX;
  _Float16* hC1 = hQKV;
  _Float16* hC2 = hQKV + 4400000;

  auto cvt = [&](const float* in, _Float16* o, int n) {
    k_cvt<<<dim3((n + 255) / 256), dim3(256), 0, stream>>>(in, o, n);
  };
  auto gemm = [&](const _Float16* A, const _Float16* W, const float* bbp,
                  float* Cf, _Float16* C16, int m, int n, int k, int relu) {
    int tiles = (m >> 5) * (n >> 6);
    int blocks = (tiles + 7) / 8;
    k_gemm<<<dim3(blocks), dim3(256), 0, stream>>>(A, W, bbp, Cf, C16, m, n, k, relu);
  };
  auto conv = [&](const _Float16* X, const _Float16* Wt16, const float* cbp,
                  const float* g, const float* bbp, const float* mm, const float* vv,
                  float* Y, _Float16* Y16, int Cin, int Lin, int Cout, int Lout,
                  int KT, int PAD, int relu) {
    long tiles = 8L * (Cout >> 5) * ((Lout + 63) >> 6);
    int blocks = (int)((tiles + 7) / 8);
    k_conv_bn<<<dim3(blocks), dim3(256), 0, stream>>>(X, Wt16, cbp, g, bbp, mm, vv,
                                                      Y, Y16, 8, Cin, Lin, Cout,
                                                      Lout, KT, PAD, relu);
  };
  auto pool = [&](const float* X, float* Y, int BC, int Lin, int Lout) {
    int tot = BC * Lout;
    k_pool<<<dim3((tot + 255) / 256), dim3(256), 0, stream>>>(X, Y, BC, Lin, Lout);
  };

  // ---- weight conversions (f32 -> f16)
  cvt(F(27), hc1w0, 524288);  cvt(F(33), hc2w0, 524288);
  cvt(F(39), hc1w1, 262144);  cvt(F(45), hc2w1, 131072);  cvt(F(51), hscw1, 32768);
  cvt(F(57), hc1w2, 131072);  cvt(F(63), hc2w2, 131072);
  _Float16* hWL[2][4];
  for (int L = 0; L < 2; ++L) {
    int base = 3 + L * 12;
    _Float16* wl = hWt + (size_t)L * 1310720;
    hWL[L][0] = wl;               cvt(F(base + 0), hWL[L][0], 196608);  // in_proj
    hWL[L][1] = wl + 196608;      cvt(F(base + 2), hWL[L][1], 65536);   // out_proj
    hWL[L][2] = wl + 262144;      cvt(F(base + 6), hWL[L][2], 524288);  // lin1
    hWL[L][3] = wl + 786432;      cvt(F(base + 8), hWL[L][3], 524288);  // lin2
  }

  const int M = BQ * SQ;  // 16384

  // ---- embedding + posenc
  k_embed<<<dim3(16384), dim3(256), 0, stream>>>(F(0), F(1), F(2), P[0], hX);

  // ---- 2 transformer encoder layers
  for (int L = 0; L < 2; ++L) {
    int base = 3 + L * 12;
    const float *ipb = F(base + 1), *opb = F(base + 3);
    const float *ln1w = F(base + 4), *ln1b = F(base + 5);
    const float *l1b = F(base + 7), *l2b = F(base + 9);
    const float *ln2w = F(base + 10), *ln2b = F(base + 11);

    gemm(hX, hWL[L][0], ipb, nullptr, hQKV, M, 768, 256, 0);      // QKV (f16 only)
    k_attn<<<dim3(2048), dim3(128), 0, stream>>>(hQKV, hAo);
    gemm(hAo, hWL[L][1], opb, P[2], nullptr, M, 256, 256, 0);     // out proj
    k_addln<<<dim3(M), dim3(256), 0, stream>>>(P[0], P[2], ln1w, ln1b, P[0], hX);
    for (int c = 0; c < 8; ++c) {                                 // chunked FFN
      gemm(hX + (size_t)c * 2048 * 256, hWL[L][2], l1b, nullptr, hH,
           2048, 2048, 256, 1);
      gemm(hH, hWL[L][3], l2b, P[2] + (size_t)c * 2048 * 256, nullptr,
           2048, 256, 2048, 0);
    }
    k_addln<<<dim3(M), dim3(256), 0, stream>>>(P[0], P[2], ln2w, ln2b, P[0], hX);
  }

  // ---- [B,S,D] -> [B,D,S]
  k_transpose<<<dim3(16384), dim3(256), 0, stream>>>(P[0], P[1], hC0);

  // ---- CNN block 0: 256->256, L=2048
  conv(hC0, hc1w0, F(28), F(29), F(30), F(31), F(32), P[2], hC1, 256, 2048, 256, 2049, 8, 4, 1);
  conv(hC1, hc2w0, F(34), F(35), F(36), F(37), F(38), P[3], nullptr, 256, 2049, 256, 2050, 8, 4, 0);
  pool(P[3], P[4], 8 * 256, 2050, 1025);
  pool(P[1], P[5], 8 * 256, 2048, 1024);
  k_addrelu<<<dim3((8*256*1024 + 255)/256), dim3(256), 0, stream>>>(P[4], 1025, P[5], 1024, P[6], hC2, 8*256, 1024);

  // ---- CNN block 1: 256->128, L=1024 (1x1 conv + BN shortcut)
  conv(hC2, hc1w1, F(40), F(41), F(42), F(43), F(44), P[1], hC0, 256, 1024, 128, 1025, 8, 4, 1);
  conv(hC0, hc2w1, F(46), F(47), F(48), F(49), F(50), P[2], nullptr, 128, 1025, 128, 1026, 8, 4, 0);
  pool(P[2], P[3], 8 * 128, 1026, 513);
  conv(hC2, hscw1, F(52), F(53), F(54), F(55), F(56), P[4], nullptr, 256, 1024, 128, 1024, 1, 0, 0);
  pool(P[4], P[5], 8 * 128, 1024, 512);
  k_addrelu<<<dim3((8*128*512 + 255)/256), dim3(256), 0, stream>>>(P[3], 513, P[5], 512, P[0], hC1, 8*128, 512);

  // ---- CNN block 2: 128->128, L=512
  conv(hC1, hc1w2, F(58), F(59), F(60), F(61), F(62), P[1], hC0, 128, 512, 128, 513, 8, 4, 1);
  conv(hC0, hc2w2, F(64), F(65), F(66), F(67), F(68), P[2], nullptr, 128, 513, 128, 514, 8, 4, 0);
  pool(P[2], P[3], 8 * 128, 514, 257);
  pool(P[0], P[4], 8 * 128, 512, 256);
  k_addrelu<<<dim3((8*128*256 + 255)/256), dim3(256), 0, stream>>>(P[3], 257, P[4], 256, P[5], nullptr, 8*128, 256);

  // ---- global average pool + classifier
  k_mean<<<dim3(4), dim3(256), 0, stream>>>(P[5], P[6], 8 * 128, 256);
  k_fc<<<dim3(8), dim3(256), 0, stream>>>(P[6], F(69), F(70), P[1], 8, 256, 128, 1);
  k_fc<<<dim3(4), dim3(256), 0, stream>>>(P[1], F(71), F(72), P[2], 8, 128, 256, 1);
  k_fc<<<dim3(1), dim3(256), 0, stream>>>(P[2], F(73), F(74), (float*)d_out, 8, 2, 128, 0);
}
